// SelectiveScanMamba_23957327577740
// MI455X (gfx1250) — compile-verified, matrix operand order checked by reading, not run on hardware
//
#include <hip/hip_runtime.h>
#include <hip/hip_bf16.h>

// ---------------------------------------------------------------------------
// Mamba selective-scan block for MI455X (gfx1250, wave32, WMMA + async-LDS).
//   prep) bf16 copies: Xb = bf16(X); Wt_* = bf16(W_*)^T  (K-major operands)
//   1) xr   = X @ W_in          (WMMA bf16 GEMM, async-to-LDS double buffered)
//   2) xi   = silu(conv(xr[:, :2048]))   (writes f32 + bf16)
//   3) dtp  = xi @ W_dt         (WMMA GEMM)
//   4) BC   = xi @ W_xproj      (N=32 dot kernel)
//   5) delta= softplus(dtp + b_dt)
//   6) yg   = gate(scan(...) + xi*D, silu(res))          (writes bf16)
//   7) out  = yg @ W_out        (WMMA GEMM)
// ---------------------------------------------------------------------------

#define BATCH   2
#define LENGTH  2048
#define D_MODEL 1024
#define D_STATE 16
#define D_CONV  4
#define D_INNER 2048
#define ROWS    (BATCH * LENGTH)          // 4096

typedef __attribute__((ext_vector_type(16))) __bf16 v16bf;
typedef __attribute__((ext_vector_type(8)))  float  v8f;

union Frag16 {
    v16bf v;
    unsigned int u[8];
};

__device__ __forceinline__ unsigned short f2bf(float x) {
    unsigned int u = __float_as_uint(x);
    unsigned int r = u + 0x7FFFu + ((u >> 16) & 1u);   // round-to-nearest-even
    return (unsigned short)(r >> 16);
}

__device__ __forceinline__ float silu_f(float x) {
    return x / (1.0f + __expf(-x));
}

// Low 32 bits of a flat pointer into LDS == byte offset within the group's
// LDS allocation (ISA 10.2: LDS aperture keeps the offset in addr[31:0]).
__device__ __forceinline__ unsigned lds_lo32(const void* p) {
    return (unsigned)(unsigned long long)p;
}

// ---------------------------------------------------------------------------
// WMMA bf16 GEMM:  C[M,N](f32) = A[M,K](bf16) * Bt[N,K](bf16, pre-transposed)
// Block 256 threads = 8 waves (4Mx2N), block tile 256x128, wave tile 64x64,
// K step 32.  Tiles staged with GLOBAL_LOAD_ASYNC_TO_LDS_B64 (ASYNCcnt),
// double buffered.  Requires M%256==0, N%128==0, K%32==0 (true here).
// ---------------------------------------------------------------------------
#define LSTR 40   // LDS row stride in ushorts (80B: 8B-aligned, conflict-free)

__device__ __forceinline__ void stage_tiles_async(
    const unsigned short* __restrict__ A, const unsigned short* __restrict__ Bt,
    int K, int bm, int bn, int k0, unsigned laBase, unsigned lbBase, int tid) {
    // A tile: 256 rows x 32 bf16, 8 B64 copies per row -> 2048 total, 8/thread
#pragma unroll
    for (int r = 0; r < 8; ++r) {
        const int idx = tid + r * 256;
        const int row = idx >> 3;
        const int s4  = (idx & 7) << 2;                 // bf16 offset in row
        const unsigned goff = (unsigned)(((unsigned)(bm + row) * (unsigned)K
                                         + (unsigned)(k0 + s4)) * 2u);
        const unsigned loff = laBase + (unsigned)((row * LSTR + s4) * 2);
        asm volatile("global_load_async_to_lds_b64 %0, %1, %2"
                     :: "v"(loff), "v"(goff), "s"(A) : "memory");
    }
    // B tile: 128 rows x 32 bf16 -> 1024 B64 copies, 4/thread
#pragma unroll
    for (int r = 0; r < 4; ++r) {
        const int idx = tid + r * 256;
        const int row = idx >> 3;
        const int s4  = (idx & 7) << 2;
        const unsigned goff = (unsigned)(((unsigned)(bn + row) * (unsigned)K
                                         + (unsigned)(k0 + s4)) * 2u);
        const unsigned loff = lbBase + (unsigned)((row * LSTR + s4) * 2);
        asm volatile("global_load_async_to_lds_b64 %0, %1, %2"
                     :: "v"(loff), "v"(goff), "s"(Bt) : "memory");
    }
}

__global__ __launch_bounds__(256)
void gemm_bf16_wmma(const unsigned short* __restrict__ A,
                    const unsigned short* __restrict__ Bt,
                    float* __restrict__ C, int M, int N, int K) {
    __shared__ __align__(16) unsigned short lA[2][256 * LSTR];   // 40 KB
    __shared__ __align__(16) unsigned short lB[2][128 * LSTR];   // 20 KB

    const int tid  = threadIdx.x;
    const int wave = tid >> 5;
    const int lane = tid & 31;
    const int bm   = blockIdx.y * 256;
    const int bn   = blockIdx.x * 128;
    const int wm   = (wave >> 1) * 64;          // 0,64,128,192
    const int wn   = (wave & 1) * 64;           // 0,64
    const int ln   = lane & 15;
    const int lh   = lane >> 4;
    const int kb   = lh * 8;

    v8f acc[4][4];
#pragma unroll
    for (int i = 0; i < 4; ++i)
#pragma unroll
        for (int j = 0; j < 4; ++j)
            acc[i][j] = (v8f){0.f, 0.f, 0.f, 0.f, 0.f, 0.f, 0.f, 0.f};

    // prologue: stage K-step 0 into buffer 0
    stage_tiles_async(A, Bt, K, bm, bn, 0, lds_lo32(&lA[0][0]),
                      lds_lo32(&lB[0][0]), tid);

    for (int k0 = 0; k0 < K; k0 += 32) {
        const int cur = (k0 >> 5) & 1;

        // copies into `cur` (issued one iteration ago) complete + visible
        asm volatile("s_wait_asynccnt 0x0" ::: "memory");
        __syncthreads();

        // overlap: stage next K-step into the other buffer
        if (k0 + 32 < K)
            stage_tiles_async(A, Bt, K, bm, bn, k0 + 32,
                              lds_lo32(&lA[cur ^ 1][0]),
                              lds_lo32(&lB[cur ^ 1][0]), tid);

        // fragments per the 16-bit A/B VGPR layout:
        // lane half 0: K = {0..7, 16..23}; half 1: K = {8..15, 24..31}
        Frag16 af[4], bfrag[4];
#pragma unroll
        for (int i = 0; i < 4; ++i) {
            const unsigned short* p = &lA[cur][(wm + i * 16 + ln) * LSTR];
#pragma unroll
            for (int j = 0; j < 4; ++j) {
                af[i].u[j]     = *(const unsigned int*)(p + kb + 2 * j);
                af[i].u[4 + j] = *(const unsigned int*)(p + 16 + kb + 2 * j);
            }
        }
#pragma unroll
        for (int j = 0; j < 4; ++j) {
            const unsigned short* p = &lB[cur][(wn + j * 16 + ln) * LSTR];
#pragma unroll
            for (int q = 0; q < 4; ++q) {
                bfrag[j].u[q]     = *(const unsigned int*)(p + kb + 2 * q);
                bfrag[j].u[4 + q] = *(const unsigned int*)(p + 16 + kb + 2 * q);
            }
        }

#pragma unroll
        for (int i = 0; i < 4; ++i)
#pragma unroll
            for (int j = 0; j < 4; ++j)
                acc[i][j] = __builtin_amdgcn_wmma_f32_16x16x32_bf16(
                    false, af[i].v, false, bfrag[j].v,
                    (short)0, acc[i][j], false, false);

        __syncthreads();   // all waves done reading `cur` before it is refilled
    }

    // store D: VGPR r holds M = r + 8*lanehalf, N = ln
#pragma unroll
    for (int i = 0; i < 4; ++i)
#pragma unroll
        for (int j = 0; j < 4; ++j) {
            const int n = bn + wn + j * 16 + ln;
#pragma unroll
            for (int r = 0; r < 8; ++r) {
                const int m = bm + wm + i * 16 + r + 8 * lh;
                C[(size_t)m * N + n] = acc[i][j][r];
            }
        }
}

// ---------------------------------------------------------------------------
// f32 -> bf16, same layout (for X)
// ---------------------------------------------------------------------------
__global__ __launch_bounds__(256)
void cvt_bf16_kernel(const float* __restrict__ src, unsigned short* __restrict__ dst,
                     size_t n) {
    const size_t g = (size_t)blockIdx.x * 256 + threadIdx.x;
    if (g < n) dst[g] = f2bf(src[g]);
}

// ---------------------------------------------------------------------------
// W[K][N] f32 -> Wt[N][K] bf16, 32x32 LDS-tiled transpose (coalesced both ways)
// ---------------------------------------------------------------------------
__global__ __launch_bounds__(256)
void transpose_cvt_kernel(const float* __restrict__ W, unsigned short* __restrict__ Wt,
                          int K, int N) {
    __shared__ unsigned short t[32][33];
    const int tx = threadIdx.x & 31;
    const int ty = threadIdx.x >> 5;            // 0..7
    const int k0 = blockIdx.y * 32;
    const int n0 = blockIdx.x * 32;
#pragma unroll
    for (int r = 0; r < 4; ++r)
        t[ty + 8 * r][tx] = f2bf(W[(size_t)(k0 + ty + 8 * r) * N + n0 + tx]);
    __syncthreads();
#pragma unroll
    for (int r = 0; r < 4; ++r)
        Wt[(size_t)(n0 + ty + 8 * r) * K + k0 + tx] = t[tx][ty + 8 * r];
}

// ---------------------------------------------------------------------------
// Causal depthwise conv (k=4, left pad 3) + SiLU; writes f32 and bf16 copies.
// ---------------------------------------------------------------------------
__global__ __launch_bounds__(256)
void conv_silu_kernel(const float* __restrict__ xr, const float* __restrict__ cw,
                      const float* __restrict__ cb, float* __restrict__ xi,
                      unsigned short* __restrict__ xib) {
    const size_t g = (size_t)blockIdx.x * 256 + threadIdx.x;
    if (g >= (size_t)ROWS * D_INNER) return;
    const int d = (int)(g % D_INNER);
    const int l = (int)((g / D_INNER) % LENGTH);
    const int b = (int)(g / ((size_t)D_INNER * LENGTH));

    float acc = cb[d];
#pragma unroll
    for (int j = 0; j < D_CONV; ++j) {
        const int ls = l - (D_CONV - 1) + j;
        if (ls >= 0)
            acc += xr[((size_t)b * LENGTH + ls) * (2 * D_INNER) + d] * cw[d * D_CONV + j];
    }
    const float v = silu_f(acc);
    xi[g]  = v;
    xib[g] = f2bf(v);
}

// ---------------------------------------------------------------------------
// x_proj: BC[m, n] = sum_k xi[m,k] * W_xproj[k,n], n = 0..31.
// ---------------------------------------------------------------------------
__global__ __launch_bounds__(256)
void xproj_kernel(const float* __restrict__ xi, const float* __restrict__ W,
                  float* __restrict__ BC) {
    const int n  = threadIdx.x & 31;
    const int mr = threadIdx.x >> 5;
    const int m  = blockIdx.x * 8 + mr;
    const float* xrow = xi + (size_t)m * D_INNER;
    float acc = 0.f;
    for (int k = 0; k < D_INNER; ++k)
        acc += xrow[k] * W[(size_t)k * 32 + n];
    BC[(size_t)m * 32 + n] = acc;
}

// ---------------------------------------------------------------------------
// delta = softplus(dtp + b_dt), in place.
// ---------------------------------------------------------------------------
__global__ __launch_bounds__(256)
void softplus_kernel(float* __restrict__ dtp, const float* __restrict__ b_dt) {
    const size_t g = (size_t)blockIdx.x * 256 + threadIdx.x;
    if (g >= (size_t)ROWS * D_INNER) return;
    const float v = dtp[g] + b_dt[g % D_INNER];
    dtp[g] = (v > 20.0f) ? v : log1pf(__expf(v));
}

// ---------------------------------------------------------------------------
// Selective scan + skip + gate.  One thread per (b, d) channel; 16 states in
// registers; B/C staged through LDS in 64-step chunks.  Writes bf16.
// ---------------------------------------------------------------------------
#define SCAN_CHUNK 64

__global__ __launch_bounds__(256)
void scan_kernel(const float* __restrict__ xi, const float* __restrict__ delta,
                 const float* __restrict__ BC, const float* __restrict__ A_log,
                 const float* __restrict__ D_param, const float* __restrict__ xr,
                 unsigned short* __restrict__ yg) {
    __shared__ float sBC[SCAN_CHUNK * 2 * D_STATE];   // 8 KB

    const int tid = threadIdx.x;
    const int d   = blockIdx.x * 256 + tid;
    const int b   = blockIdx.y;

    float A[D_STATE];
#pragma unroll
    for (int n = 0; n < D_STATE; ++n) A[n] = -__expf(A_log[n]);
    const float Dp = D_param[d];

    float h[D_STATE];
#pragma unroll
    for (int n = 0; n < D_STATE; ++n) h[n] = 0.f;

    for (int c = 0; c < LENGTH / SCAN_CHUNK; ++c) {
        __syncthreads();
#pragma unroll
        for (int r = 0; r < (SCAN_CHUNK * 2 * D_STATE) / 256; ++r) {
            const int i = tid + r * 256;
            sBC[i] = BC[((size_t)b * LENGTH + c * SCAN_CHUNK) * (2 * D_STATE) + i];
        }
        __syncthreads();

        for (int lc = 0; lc < SCAN_CHUNK; ++lc) {
            const int l = c * SCAN_CHUNK + lc;
            const size_t off = ((size_t)b * LENGTH + l) * D_INNER + d;
            const float u  = xi[off];
            const float dl = delta[off];
            const float* Bm = &sBC[lc * 2 * D_STATE];
            const float* Cm = Bm + D_STATE;
            float y = 0.f;
#pragma unroll
            for (int n = 0; n < D_STATE; ++n) {
                const float dA = __expf(dl * A[n]);
                h[n] = h[n] * dA + (dl * Bm[n]) * u;
                y += h[n] * Cm[n];
            }
            const float res = xr[((size_t)b * LENGTH + l) * (2 * D_INNER) + D_INNER + d];
            yg[off] = f2bf((y + u * Dp) * silu_f(res));
        }
    }
}

// ---------------------------------------------------------------------------
// Workspace layout (bytes)
// ---------------------------------------------------------------------------
#define WS_XR    ((size_t)0)                                    // f32 4096x4096
#define WS_XI    (WS_XR   + (size_t)ROWS * 2 * D_INNER * 4)     // f32 4096x2048
#define WS_DTP   (WS_XI   + (size_t)ROWS * D_INNER * 4)         // f32 4096x2048
#define WS_BC    (WS_DTP  + (size_t)ROWS * D_INNER * 4)         // f32 4096x32
#define WS_XB    (WS_BC   + (size_t)ROWS * 2 * D_STATE * 4)     // bf16 4096x1024
#define WS_WTIN  (WS_XB   + (size_t)ROWS * D_MODEL * 2)         // bf16 4096x1024
#define WS_XIB   (WS_WTIN + (size_t)(2 * D_INNER) * D_MODEL * 2)// bf16 4096x2048
#define WS_WTDT  (WS_XIB  + (size_t)ROWS * D_INNER * 2)         // bf16 2048x2048
#define WS_WTOUT (WS_WTDT + (size_t)D_INNER * D_INNER * 2)      // bf16 1024x2048
#define WS_YGB   (WS_WTOUT+ (size_t)D_MODEL * D_INNER * 2)      // bf16 4096x2048

extern "C" void kernel_launch(void* const* d_in, const int* in_sizes, int n_in,
                              void* d_out, int out_size, void* d_ws, size_t ws_size,
                              hipStream_t stream) {
    const float* x      = (const float*)d_in[0];
    const float* W_in   = (const float*)d_in[1];
    const float* conv_w = (const float*)d_in[2];
    const float* conv_b = (const float*)d_in[3];
    const float* W_xprj = (const float*)d_in[4];
    const float* W_dt   = (const float*)d_in[5];
    const float* b_dt   = (const float*)d_in[6];
    const float* A_log  = (const float*)d_in[7];
    const float* D_par  = (const float*)d_in[8];
    const float* W_out  = (const float*)d_in[9];
    float* out = (float*)d_out;

    char* ws = (char*)d_ws;
    float*          xr    = (float*)(ws + WS_XR);
    float*          xi    = (float*)(ws + WS_XI);
    float*          dtp   = (float*)(ws + WS_DTP);
    float*          BC    = (float*)(ws + WS_BC);
    unsigned short* Xb    = (unsigned short*)(ws + WS_XB);
    unsigned short* WtIn  = (unsigned short*)(ws + WS_WTIN);
    unsigned short* xib   = (unsigned short*)(ws + WS_XIB);
    unsigned short* WtDt  = (unsigned short*)(ws + WS_WTDT);
    unsigned short* WtOut = (unsigned short*)(ws + WS_WTOUT);
    unsigned short* ygb   = (unsigned short*)(ws + WS_YGB);

    // prep: bf16 operands (K-major on both sides)
    {
        const size_t n = (size_t)ROWS * D_MODEL;
        cvt_bf16_kernel<<<(unsigned)((n + 255) / 256), 256, 0, stream>>>(x, Xb, n);
    }
    transpose_cvt_kernel<<<dim3((2 * D_INNER) / 32, D_MODEL / 32), 256, 0, stream>>>(
        W_in, WtIn, D_MODEL, 2 * D_INNER);
    transpose_cvt_kernel<<<dim3(D_INNER / 32, D_INNER / 32), 256, 0, stream>>>(
        W_dt, WtDt, D_INNER, D_INNER);
    transpose_cvt_kernel<<<dim3(D_MODEL / 32, D_INNER / 32), 256, 0, stream>>>(
        W_out, WtOut, D_INNER, D_MODEL);

    // 1) xr = X @ W_in : [4096,1024] x [1024,4096]
    gemm_bf16_wmma<<<dim3((2 * D_INNER) / 128, ROWS / 256), 256, 0, stream>>>(
        Xb, WtIn, xr, ROWS, 2 * D_INNER, D_MODEL);

    // 2) xi = silu(conv(xr[:, :D_INNER]))
    {
        const size_t n = (size_t)ROWS * D_INNER;
        conv_silu_kernel<<<(unsigned)((n + 255) / 256), 256, 0, stream>>>(
            xr, conv_w, conv_b, xi, xib);
    }

    // 3) dtp = xi @ W_dt : [4096,2048] x [2048,2048]
    gemm_bf16_wmma<<<dim3(D_INNER / 128, ROWS / 256), 256, 0, stream>>>(
        xib, WtDt, dtp, ROWS, D_INNER, D_INNER);

    // 4) BC = xi @ W_xproj : [4096,2048] x [2048,32]
    xproj_kernel<<<ROWS / 8, 256, 0, stream>>>(xi, W_xprj, BC);

    // 5) delta = softplus(dtp + b_dt)
    {
        const size_t n = (size_t)ROWS * D_INNER;
        softplus_kernel<<<(unsigned)((n + 255) / 256), 256, 0, stream>>>(dtp, b_dt);
    }

    // 6) selective scan + skip + gate -> bf16
    scan_kernel<<<dim3(D_INNER / 256, BATCH), 256, 0, stream>>>(
        xi, dtp, BC, A_log, D_par, xr, ygb);

    // 7) out = yg @ W_out : [4096,2048] x [2048,1024]
    gemm_bf16_wmma<<<dim3(D_MODEL / 128, ROWS / 256), 256, 0, stream>>>(
        ygb, WtOut, out, ROWS, D_MODEL, D_INNER);
}
